// Mamba3DFeatureExtractor_3006477107874
// MI455X (gfx1250) — matrix operand order repaired and checked by reading, not compiled, not run on hardware
//
#include <hip/hip_runtime.h>
#include <math.h>

// ---------------- model constants ----------------
constexpr int BB    = 4;
constexpr int NPTS  = 4096;
constexpr int INCH  = 6;
constexpr int DIM   = 128;
constexpr int KNB   = 16;     // KNN neighbors
constexpr int DST   = 16;     // d_state
constexpr int DIN   = 256;    // d_inner
constexpr int DTR   = 8;      // dt_rank
constexpr int ANUM  = 64;
constexpr int ROWS  = BB * NPTS;          // 16384
constexpr int XPN   = DTR + 2 * DST;      // 40

// ---------------- WMMA types ----------------
typedef _Float16 f16;
typedef __attribute__((ext_vector_type(16))) _Float16 v16h;
typedef __attribute__((ext_vector_type(8)))  float    v8f;

__device__ __forceinline__ float siluf(float x) { return x / (1.f + __expf(-x)); }

// load one 16x32 A fragment for this lane (row = r16, K base = k)
__device__ __forceinline__ v16h load_afrag(const float* __restrict__ Ar, int k, int hi)
{
    const float4* __restrict__ ap = (const float4*)(Ar + k + (hi << 4));
    const float4 a0 = ap[0];   // ka+0..3
    const float4 a1 = ap[1];   // ka+4..7
    const float4 a2 = ap[2];   // ka+8..11
    const float4 a3 = ap[3];   // ka+12..15
    v16h f;
    f[0]  = (f16)a0.x; f[1]  = (f16)a0.y; f[2]  = (f16)a0.z; f[3]  = (f16)a0.w;
    f[4]  = (f16)a1.x; f[5]  = (f16)a1.y; f[6]  = (f16)a1.z; f[7]  = (f16)a1.w;
    f[8]  = (f16)a2.x; f[9]  = (f16)a2.y; f[10] = (f16)a2.z; f[11] = (f16)a2.w;
    f[12] = (f16)a3.x; f[13] = (f16)a3.y; f[14] = (f16)a3.z; f[15] = (f16)a3.w;
    return f;
}

// load one 32x16 B fragment; NCOLS is compile-time so the 16 stride-NCOLS
// loads become immediate-offset global_load_b32 from a single base address.
template <int NCOLS>
__device__ __forceinline__ v16h load_bfrag(const float* __restrict__ Wp)
{
    v16h f;
#pragma unroll
    for (int e = 0; e < 16; ++e) f[e] = (f16)Wp[(size_t)e * NCOLS];
    return f;
}

// =====================================================================
// WMMA GEMM:  C[M,NCOLS] = act( gather(A)[M,Kd] @ W[Kd,NCOLS] + bias + res )
// REQUIREMENTS (all call sites satisfy them):
//   M % 32 == 0, Kd % 32 == 0, lda % 4 == 0, NCOLS = weight leading dim.
// One wave computes a 32x32 macro-tile = 2x2 WMMA tiles: per K-step, 2 A
// fragments + 2 B fragments feed 4 v_wmma ops. NCOLS is a template constant
// so B-loads use immediate offsets (no per-element 64-bit address math).
// Out-of-range N columns are address-clamped; their lanes' stores are masked.
// =====================================================================
template <int NCOLS>
__global__ void __launch_bounds__(128, 1)
k_gemm(const float* __restrict__ A, int lda,
       const float* __restrict__ W,
       const float* __restrict__ bias,
       const float* __restrict__ res, int ldres,
       float* __restrict__ C, int ldc,
       const int* __restrict__ rowIdx,
       int M, int Kd, int act)
{
    const int lane = threadIdx.x & 31;
    const int wid  = threadIdx.x >> 5;
    constexpr int ntn = (NCOLS + 31) >> 5;      // 32-wide super-tiles
    const int ntm  = M >> 5;
    const int tile = blockIdx.x * (blockDim.x >> 5) + wid;
    if (tile >= ntm * ntn) return;              // wave-uniform
    const int tm = tile / ntn;
    const int tn = tile % ntn;

    const int r16 = lane & 15;
    const int hi  = lane >> 4;                  // 0 or 1 (upper lane half)

    const int rowA0 = tm * 32 + r16;            // < M  (M % 32 == 0)
    const int rowA1 = rowA0 + 16;               // < M
    const int colB0 = tn * 32 + r16;
    const int colB1 = colB0 + 16;
    const int colS0 = (colB0 < NCOLS) ? colB0 : (NCOLS - 1);
    const int colS1 = (colB1 < NCOLS) ? colB1 : (NCOLS - 1);
    const int arow0 = rowIdx ? rowIdx[rowA0] : rowA0;
    const int arow1 = rowIdx ? rowIdx[rowA1] : rowA1;
    const float* __restrict__ Ar0 = A + (size_t)arow0 * lda;
    const float* __restrict__ Ar1 = A + (size_t)arow1 * lda;
    // B base pointers include the per-half-wave K offset (hi*16 rows)
    const float* __restrict__ Wp0 = W + (size_t)(hi << 4) * NCOLS + colS0;
    const float* __restrict__ Wp1 = W + (size_t)(hi << 4) * NCOLS + colS1;

    v8f acc00 = {0.f,0.f,0.f,0.f,0.f,0.f,0.f,0.f};
    v8f acc01 = acc00, acc10 = acc00, acc11 = acc00;

    for (int k0 = 0; k0 < Kd; k0 += 32) {
        const v16h a0 = load_afrag(Ar0, k0, hi);
        const v16h a1 = load_afrag(Ar1, k0, hi);
        const v16h b0 = load_bfrag<NCOLS>(Wp0);
        const v16h b1 = load_bfrag<NCOLS>(Wp1);
        Wp0 += 32 * NCOLS;
        Wp1 += 32 * NCOLS;
        acc00 = __builtin_amdgcn_wmma_f32_16x16x32_f16(false, a0, false, b0, (short)0, acc00, false, false);
        acc01 = __builtin_amdgcn_wmma_f32_16x16x32_f16(false, a0, false, b1, (short)0, acc01, false, false);
        acc10 = __builtin_amdgcn_wmma_f32_16x16x32_f16(false, a1, false, b0, (short)0, acc10, false, false);
        acc11 = __builtin_amdgcn_wmma_f32_16x16x32_f16(false, a1, false, b1, (short)0, acc11, false, false);
    }

    // epilogue: C/D layout -> lane col N = r16 (+16 for *1 tiles),
    // vgpr v -> row M = v + 8*hi (+16 for *1* row tiles)
#pragma unroll
    for (int tj = 0; tj < 2; ++tj) {
        const int cn = tn * 32 + tj * 16 + r16;
        if (cn >= NCOLS) continue;
        const float bv = bias ? bias[cn] : 0.f;
#pragma unroll
        for (int ti = 0; ti < 2; ++ti) {
            const v8f accv = (ti == 0) ? (tj == 0 ? acc00 : acc01)
                                       : (tj == 0 ? acc10 : acc11);
#pragma unroll
            for (int vv = 0; vv < 8; ++vv) {
                const int cm = tm * 32 + ti * 16 + vv + (hi << 3);
                float val = accv[vv] + bv;
                if (res) val += res[(size_t)cm * ldres + cn];
                if (act == 1) val = siluf(val);
                C[(size_t)cm * ldc + cn] = val;
            }
        }
    }
}

// =====================================================================
// Embedding: h = x@glob_w + glob_b + (silu(xyz@pos_w1+b1) @ pos_w2 + b2)
// one block (128 thr) per point, hidden staged in LDS.
// =====================================================================
__global__ void k_embed(const float* __restrict__ x,
                        const float* __restrict__ gw, const float* __restrict__ gb,
                        const float* __restrict__ pw1, const float* __restrict__ pb1,
                        const float* __restrict__ pw2, const float* __restrict__ pb2,
                        float* __restrict__ hout)
{
    const int row = blockIdx.x;
    const int d = threadIdx.x;
    __shared__ float hid[DIM];
    const float* xr = x + (size_t)row * INCH;
    float s = pb1[d];
#pragma unroll
    for (int j = 0; j < 3; ++j) s += xr[j] * pw1[j * DIM + d];
    hid[d] = siluf(s);
    __syncthreads();
    float o = gb[d];
#pragma unroll
    for (int j = 0; j < 6; ++j) o += xr[j] * gw[j * DIM + d];
    float p = pb2[d];
    for (int kx = 0; kx < DIM; ++kx) p += hid[kx] * pw2[kx * DIM + d];
    hout[(size_t)row * DIM + d] = o + p;
}

// =====================================================================
// LayerNorm over 128 channels (optionally fused SiLU). block = row.
// =====================================================================
__global__ void k_layernorm(const float* __restrict__ in, int ldin,
                            float* __restrict__ out, int ldout,
                            const float* __restrict__ g, const float* __restrict__ b,
                            int act, int rows)
{
    const int row = blockIdx.x;
    if (row >= rows) return;
    const int d = threadIdx.x;
    __shared__ float red[DIM];
    const float v = in[(size_t)row * ldin + d];
    red[d] = v;
    __syncthreads();
    for (int s = 64; s > 0; s >>= 1) { if (d < s) red[d] += red[d + s]; __syncthreads(); }
    const float mu = red[0] * (1.f / DIM);
    __syncthreads();
    const float c = v - mu;
    red[d] = c * c;
    __syncthreads();
    for (int s = 64; s > 0; s >>= 1) { if (d < s) red[d] += red[d + s]; __syncthreads(); }
    const float var = red[0] * (1.f / DIM);
    float o = c * rsqrtf(var + 1e-5f) * g[d] + b[d];
    if (act == 1) o = siluf(o);
    out[(size_t)row * ldout + d] = o;
}

// =====================================================================
// Brute-force KNN (K=16 smallest squared distances, ascending, self incl.)
// one thread per query.
// =====================================================================
__global__ void k_knn(const float* __restrict__ x, int* __restrict__ idx)
{
    const int t = blockIdx.x * blockDim.x + threadIdx.x;
    if (t >= ROWS) return;
    const int b = t / NPTS, n = t % NPTS;
    const float* base = x + (size_t)b * NPTS * INCH;
    const float qx = base[n * INCH + 0], qy = base[n * INCH + 1], qz = base[n * INCH + 2];
    float bestd[KNB]; int besti[KNB];
#pragma unroll
    for (int i = 0; i < KNB; ++i) { bestd[i] = 1e30f; besti[i] = 0; }
    for (int m = 0; m < NPTS; ++m) {
        const float dx = base[m * INCH + 0] - qx;
        const float dy = base[m * INCH + 1] - qy;
        const float dz = base[m * INCH + 2] - qz;
        const float dd = dx * dx + dy * dy + dz * dz;
        if (dd < bestd[KNB - 1]) {
            int j = KNB - 1;
            while (j > 0 && bestd[j - 1] > dd) {
                bestd[j] = bestd[j - 1]; besti[j] = besti[j - 1]; --j;
            }
            bestd[j] = dd; besti[j] = m;
        }
    }
#pragma unroll
    for (int i = 0; i < KNB; ++i) idx[(size_t)t * KNB + i] = besti[i];
}

// gather-row index builders
__global__ void k_gidx_lga(const int* __restrict__ idx, int* __restrict__ gidx,
                           int qbase, int nrows)
{
    const int r = blockIdx.x * blockDim.x + threadIdx.x;
    if (r >= nrows) return;
    const int qq = qbase + r / KNB;
    const int b = qq / NPTS;
    gidx[r] = b * NPTS + idx[(size_t)qq * KNB + (r % KNB)];
}
__global__ void k_gidx_anchor(const int* __restrict__ aidx, int* __restrict__ gidx)
{
    const int t = threadIdx.x;            // 256
    gidx[t] = (t / ANUM) * NPTS + aidx[t];
}

// =====================================================================
// LGA geometry layer 1: silu(BN(rel @ geo_w1 + b1)) -> hid[rows,64]
// block = (query,neighbor) row, 64 threads.
// =====================================================================
__global__ void k_geo1(const float* __restrict__ x, const int* __restrict__ idx,
                       const float* __restrict__ w1, const float* __restrict__ b1,
                       const float* __restrict__ bng, const float* __restrict__ bnb,
                       const float* __restrict__ bnm, const float* __restrict__ bnv,
                       float* __restrict__ hid, int qbase, int nrows)
{
    const int r = blockIdx.x;
    if (r >= nrows) return;
    const int d = threadIdx.x;            // 0..63
    const int qq = qbase + r / KNB;
    const int b = qq / NPTS, n = qq % NPTS;
    const int nb = idx[(size_t)qq * KNB + (r % KNB)];
    const float* xb = x + (size_t)b * NPTS * INCH;
    const float rx = xb[nb * INCH + 0] - xb[n * INCH + 0];
    const float ry = xb[nb * INCH + 1] - xb[n * INCH + 1];
    const float rz = xb[nb * INCH + 2] - xb[n * INCH + 2];
    float gv = rx * w1[0 * 64 + d] + ry * w1[1 * 64 + d] + rz * w1[2 * 64 + d] + b1[d];
    gv = (gv - bnm[d]) * rsqrtf(bnv[d] + 1e-5f) * bng[d] + bnb[d];
    hid[(size_t)r * 64 + d] = siluf(gv);
}

// =====================================================================
// LGA aggregation: fused' = alpha*fused+beta ; w = softmax_k(sum_d fused')
// out[q] = sum_k w_k * fused'_k.   block = query, 128 threads, LDS staging.
// =====================================================================
__global__ void k_lga_agg(const float* __restrict__ fused,
                          const float* __restrict__ alpha, const float* __restrict__ beta,
                          float* __restrict__ hout, int qbase)
{
    const int q = blockIdx.x;
    const int d = threadIdx.x;
    __shared__ float vals[KNB * DIM];     // 8 KB
    __shared__ float ssum[KNB];
    __shared__ float sw[KNB];
    const float al = alpha[d], be = beta[d];
    float myv[KNB];
#pragma unroll
    for (int k = 0; k < KNB; ++k) {
        myv[k] = al * fused[((size_t)q * KNB + k) * DIM + d] + be;
        vals[k * DIM + d] = myv[k];
    }
    __syncthreads();
    if (d < KNB) {
        float s = 0.f;
        for (int j = 0; j < DIM; ++j) s += vals[d * DIM + j];
        ssum[d] = s;
    }
    __syncthreads();
    if (d == 0) {
        float mx = -1e30f;
        for (int k = 0; k < KNB; ++k) mx = fmaxf(mx, ssum[k]);
        float den = 0.f;
        for (int k = 0; k < KNB; ++k) { sw[k] = __expf(ssum[k] - mx); den += sw[k]; }
        const float inv = 1.f / den;
        for (int k = 0; k < KNB; ++k) sw[k] *= inv;
    }
    __syncthreads();
    float o = 0.f;
#pragma unroll
    for (int k = 0; k < KNB; ++k) o += sw[k] * myv[k];
    hout[(size_t)(qbase + q) * DIM + d] = o;
}

// =====================================================================
// Mamba depthwise causal conv (width 4) + SiLU; dir=1 processes reversed seq.
// =====================================================================
__global__ void k_conv(const float* __restrict__ xz, const float* __restrict__ cw,
                       const float* __restrict__ cb, float* __restrict__ xc, int dir)
{
    const int t = blockIdx.x * blockDim.x + threadIdx.x;
    if (t >= ROWS * DIN) return;
    const int c = t & (DIN - 1);
    const int r = t >> 8;
    const int n = r & (NPTS - 1);
    const int b = r >> 12;
    const float* base = xz + (size_t)(b << 12) * (2 * DIN) + c;
    float acc = cb[c];
#pragma unroll
    for (int j = 0; j < 4; ++j) {
        const int m = dir ? (n + 3 - j) : (n - 3 + j);
        if (m >= 0 && m < NPTS) acc += cw[c * 4 + j] * base[(size_t)m * (2 * DIN)];
    }
    xc[t] = siluf(acc);
}

// dt = softplus(dbl[:, :8] @ dt_w + dt_b)
__global__ void k_dt(const float* __restrict__ dbl, const float* __restrict__ dtw,
                     const float* __restrict__ dtb, float* __restrict__ dt)
{
    const int t = blockIdx.x * blockDim.x + threadIdx.x;
    if (t >= ROWS * DIN) return;
    const int c = t & (DIN - 1);
    const int r = t >> 8;
    const float* dr = dbl + (size_t)r * XPN;
    float s = dtb[c];
#pragma unroll
    for (int j = 0; j < DTR; ++j) s += dr[j] * dtw[j * DIN + c];
    dt[t] = (s > 20.f) ? s : log1pf(__expf(s));
}

// selective scan: one thread per (batch, channel), 16 states in registers
__global__ void k_scan(const float* __restrict__ xc, const float* __restrict__ dbl,
                       const float* __restrict__ dt, const float* __restrict__ Alog,
                       const float* __restrict__ Dp, float* __restrict__ y, int dir)
{
    const int t = blockIdx.x * blockDim.x + threadIdx.x;
    if (t >= BB * DIN) return;
    const int c = t & (DIN - 1);
    const int b = t >> 8;
    float a[DST], hs[DST];
#pragma unroll
    for (int s = 0; s < DST; ++s) { a[s] = -__expf(Alog[c * DST + s]); hs[s] = 0.f; }
    const float dcoef = Dp[c];
    for (int step = 0; step < NPTS; ++step) {
        const int n = dir ? (NPTS - 1 - step) : step;
        const size_t r = ((size_t)b << 12) + n;
        const float dtv = dt[r * DIN + c];
        const float xv = xc[r * DIN + c];
        const float* Bm = dbl + r * XPN + DTR;
        const float* Cm = dbl + r * XPN + DTR + DST;
        const float dx = dtv * xv;
        float yv = 0.f;
#pragma unroll
        for (int s = 0; s < DST; ++s) {
            hs[s] = hs[s] * __expf(dtv * a[s]) + dx * Bm[s];
            yv += hs[s] * Cm[s];
        }
        y[r * DIN + c] = yv + xv * dcoef;
    }
}

// yg = y * silu(z)   (z = in_proj cols 256..511)
__global__ void k_gate(const float* __restrict__ y, const float* __restrict__ xz,
                       float* __restrict__ yg)
{
    const int t = blockIdx.x * blockDim.x + threadIdx.x;
    if (t >= ROWS * DIN) return;
    const int c = t & (DIN - 1);
    const int r = t >> 8;
    const float z = xz[(size_t)r * (2 * DIN) + DIN + c];
    yg[t] = y[t] * siluf(z);
}

// =====================================================================
// Anchor MHA core: block = (batch, head); thread = query row.
// two-pass softmax, head_dim=32 kept in registers.
// =====================================================================
__global__ void k_attn(const float* __restrict__ q, const float* __restrict__ kk,
                       const float* __restrict__ v, float* __restrict__ o)
{
    const int b = blockIdx.x >> 2;
    const int hh = blockIdx.x & 3;
    const int i = threadIdx.x;            // 0..63
    const float sc = 0.17677669529663687f; // 32^-0.5
    const float* qr = q + ((size_t)(b * ANUM + i)) * DIM + hh * 32;
    float qv[32];
#pragma unroll
    for (int d = 0; d < 32; ++d) qv[d] = qr[d];
    float mx = -1e30f;
    for (int m = 0; m < ANUM; ++m) {
        const float* kr = kk + ((size_t)(b * ANUM + m)) * DIM + hh * 32;
        float s = 0.f;
#pragma unroll
        for (int d = 0; d < 32; ++d) s += qv[d] * kr[d];
        mx = fmaxf(mx, s * sc);
    }
    float den = 0.f, outv[32];
#pragma unroll
    for (int d = 0; d < 32; ++d) outv[d] = 0.f;
    for (int m = 0; m < ANUM; ++m) {
        const float* kr = kk + ((size_t)(b * ANUM + m)) * DIM + hh * 32;
        float s = 0.f;
#pragma unroll
        for (int d = 0; d < 32; ++d) s += qv[d] * kr[d];
        s = __expf(s * sc - mx);
        den += s;
        const float* vr = v + ((size_t)(b * ANUM + m)) * DIM + hh * 32;
#pragma unroll
        for (int d = 0; d < 32; ++d) outv[d] += s * vr[d];
    }
    const float inv = 1.f / den;
    float* orow = o + ((size_t)(b * ANUM + i)) * DIM + hh * 32;
#pragma unroll
    for (int d = 0; d < 32; ++d) orow[d] = outv[d] * inv;
}

// =====================================================================
// Host orchestration
// =====================================================================
static inline void gemm(hipStream_t st, const float* A, int lda, const float* W,
                        const float* bias, const float* res, int ldres,
                        float* C, int ldc, const int* rowIdx,
                        int M, int Ncols, int Kd, int act)
{
    const int tiles = (M / 32) * ((Ncols + 31) / 32);   // 32x32 macro-tiles
    dim3 blk(128);                                      // 4 waves / block
    dim3 grd((tiles + 3) / 4);
    switch (Ncols) {
    case 512:
        k_gemm<512><<<grd, blk, 0, st>>>(A, lda, W, bias, res, ldres, C, ldc, rowIdx, M, Kd, act);
        break;
    case 256:
        k_gemm<256><<<grd, blk, 0, st>>>(A, lda, W, bias, res, ldres, C, ldc, rowIdx, M, Kd, act);
        break;
    case 128:
        k_gemm<128><<<grd, blk, 0, st>>>(A, lda, W, bias, res, ldres, C, ldc, rowIdx, M, Kd, act);
        break;
    case 40:
        k_gemm<40><<<grd, blk, 0, st>>>(A, lda, W, bias, res, ldres, C, ldc, rowIdx, M, Kd, act);
        break;
    default:    // unused
        k_gemm<128><<<grd, blk, 0, st>>>(A, lda, W, bias, res, ldres, C, ldc, rowIdx, M, Kd, act);
        break;
    }
}

// flattened parameter indices (insertion order of setup_inputs dicts)
static const int PI_GLOB_W = 2, PI_GLOB_B = 3;
static const int PI_POS_W1 = 4, PI_POS_B1 = 5, PI_POS_W2 = 6, PI_POS_B2 = 7;
static const int PI_LGA    = 8;    // 2 x 18
static const int PI_MAMBA  = 44;   // 4 x 28 (fwd:9, bwd:9, fusion w/b, n1 g/b, n2 g/b, mlp w1/b1/w2/b2)
static const int PI_NORMS  = 156;  // 4 x 2
static const int PI_FINAL  = 164;  // g, b
static const int PI_ATTN   = 166;  // qw,qb,kw,kb,vw,vb,ow,ob

extern "C" void kernel_launch(void* const* d_in, const int* in_sizes, int n_in,
                              void* d_out, int out_size, void* d_ws, size_t ws_size,
                              hipStream_t stream)
{
    (void)in_sizes; (void)n_in; (void)out_size; (void)ws_size;
    const float* x = (const float*)d_in[0];
    const int* a_index = (const int*)d_in[1];
    auto P = [&](int i) { return (const float*)d_in[i]; };

    char* wsb = (char*)d_ws;
    auto WF = [&](size_t mb) { return (float*)(wsb + mb * (size_t)(1 << 20)); };
    float* bufH   = WF(0);     // 16384x128        (8 MB)
    float* bufXN  = WF(8);     // 16384x128        (8 MB)
    float* bufT0  = WF(16);    // xz / mlp-hid / lga-concat (32 MB)
    float* bufT1  = WF(48);    // xc / lga tf,fused (16 MB)
    float* bufDBL = WF(64);    // 16384x40          (4 MB)
    float* bufDT  = WF(68);    // 16384x256        (16 MB)
    float* bufY   = WF(84);    // 16384x256        (16 MB)
    float* bufYG  = WF(100);   // 16384x256        (16 MB)
    float* bufT6  = WF(116);   // xf|xb concat / lga h_out (16 MB)
    int*   bufIDX = (int*)(wsb + 132 * (size_t)(1 << 20));   // KNN idx (1 MB)
    float* bufGH  = WF(133);   // geo hidden chunk  (8 MB)
    int*   bufGIDX= (int*)(wsb + 141 * (size_t)(1 << 20));   // gather rows (1 MB)
    float* bufQ   = WF(142);
    float* bufK   = (float*)((char*)bufQ + (size_t)256 * DIM * 4);
    float* bufV   = (float*)((char*)bufK + (size_t)256 * DIM * 4);
    float* bufAO  = (float*)((char*)bufV + (size_t)256 * DIM * 4);

    // ---- embedding + KNN (shared by both LGA layers) ----
    k_embed<<<ROWS, DIM, 0, stream>>>(x, P(PI_GLOB_W), P(PI_GLOB_B),
                                      P(PI_POS_W1), P(PI_POS_B1),
                                      P(PI_POS_W2), P(PI_POS_B2), bufH);
    k_knn<<<(ROWS + 127) / 128, 128, 0, stream>>>(x, bufIDX);

    auto lga_layer = [&](int li) {
        const int g0 = PI_LGA + li * 18;
        const int CHQ = 2048;                    // queries per chunk
        for (int ch = 0; ch < ROWS / CHQ; ++ch) {
            const int qbase = ch * CHQ;
            const int nrows = CHQ * KNB;         // 32768
            k_gidx_lga<<<(nrows + 255) / 256, 256, 0, stream>>>(bufIDX, bufGIDX, qbase, nrows);
            k_geo1<<<nrows, 64, 0, stream>>>(x, bufIDX, P(g0 + 4), P(g0 + 5),
                                             P(g0 + 6), P(g0 + 7), P(g0 + 8), P(g0 + 9),
                                             bufGH, qbase, nrows);
            // tf = silu(LN(gather(h) @ ft_w + ft_b)) -> concat cols 0:128
            gemm(stream, bufH, DIM, P(g0 + 0), P(g0 + 1), nullptr, 0,
                 bufT1, DIM, bufGIDX, nrows, DIM, DIM, 0);
            k_layernorm<<<nrows, DIM, 0, stream>>>(bufT1, DIM, bufT0, 2 * DIM,
                                                   P(g0 + 2), P(g0 + 3), 1, nrows);
            // geo = hid @ geo_w2 + b2 -> concat cols 128:256
            gemm(stream, bufGH, 64, P(g0 + 10), P(g0 + 11), nullptr, 0,
                 bufT0 + DIM, 2 * DIM, nullptr, nrows, DIM, 64, 0);
            // fused = silu(LN(concat @ fu_w + fu_b))
            gemm(stream, bufT0, 2 * DIM, P(g0 + 12), P(g0 + 13), nullptr, 0,
                 bufT1, DIM, nullptr, nrows, DIM, 2 * DIM, 0);
            k_layernorm<<<nrows, DIM, 0, stream>>>(bufT1, DIM, bufT1, DIM,
                                                   P(g0 + 14), P(g0 + 15), 1, nrows);
            k_lga_agg<<<CHQ, DIM, 0, stream>>>(bufT1, P(g0 + 16), P(g0 + 17), bufT6, qbase);
        }
        hipMemcpyAsync(bufH, bufT6, (size_t)ROWS * DIM * 4,
                       hipMemcpyDeviceToDevice, stream);
    };

    auto mamba_block = [&](int mi) {
        const int mb = PI_MAMBA + mi * 28;
        // norm1
        k_layernorm<<<ROWS, DIM, 0, stream>>>(bufH, DIM, bufXN, DIM,
                                              P(mb + 20), P(mb + 21), 0, ROWS);
        for (int dir = 0; dir < 2; ++dir) {
            const int mm = mb + (dir ? 9 : 0);
            // in_proj (per-direction weights; pointwise so no flip needed)
            gemm(stream, bufXN, DIM, P(mm + 0), nullptr, nullptr, 0,
                 bufT0, 2 * DIN, nullptr, ROWS, 2 * DIN, DIM, 0);
            k_conv<<<(ROWS * DIN + 255) / 256, 256, 0, stream>>>(bufT0, P(mm + 1), P(mm + 2),
                                                                 bufT1, dir);
            gemm(stream, bufT1, DIN, P(mm + 3), nullptr, nullptr, 0,
                 bufDBL, XPN, nullptr, ROWS, XPN, DIN, 0);
            k_dt<<<(ROWS * DIN + 255) / 256, 256, 0, stream>>>(bufDBL, P(mm + 4), P(mm + 5), bufDT);
            k_scan<<<(BB * DIN + 127) / 128, 128, 0, stream>>>(bufT1, bufDBL, bufDT,
                                                               P(mm + 6), P(mm + 7), bufY, dir);
            k_gate<<<(ROWS * DIN + 255) / 256, 256, 0, stream>>>(bufY, bufT0, bufYG);
            gemm(stream, bufYG, DIN, P(mm + 8), nullptr, nullptr, 0,
                 bufT6 + dir * DIM, 2 * DIM, nullptr, ROWS, DIM, DIN, 0);
        }
        // fusion + residual
        gemm(stream, bufT6, 2 * DIM, P(mb + 18), P(mb + 19), bufH, DIM,
             bufH, DIM, nullptr, ROWS, DIM, 2 * DIM, 0);
        // norm2 + MLP + residual
        k_layernorm<<<ROWS, DIM, 0, stream>>>(bufH, DIM, bufXN, DIM,
                                              P(mb + 22), P(mb + 23), 0, ROWS);
        gemm(stream, bufXN, DIM, P(mb + 24), P(mb + 25), nullptr, 0,
             bufT0, 4 * DIM, nullptr, ROWS, 4 * DIM, DIM, 1);
        gemm(stream, bufT0, 4 * DIM, P(mb + 26), P(mb + 27), bufH, DIM,
             bufH, DIM, nullptr, ROWS, DIM, 4 * DIM, 0);
    };

    for (int i = 0; i < 4; ++i) {
        if ((i & 1) == 0) lga_layer(i / 2);
        mamba_block(i);
        k_layernorm<<<ROWS, DIM, 0, stream>>>(bufH, DIM, bufH, DIM,
                                              P(PI_NORMS + 2 * i), P(PI_NORMS + 2 * i + 1),
                                              0, ROWS);
    }

    // final LN straight into d_out h-region
    float* outH = (float*)d_out;
    float* outA = outH + (size_t)ROWS * DIM;
    k_layernorm<<<ROWS, DIM, 0, stream>>>(bufH, DIM, outH, DIM,
                                          P(PI_FINAL), P(PI_FINAL + 1), 0, ROWS);

    // anchor MHA (gather q/k/v straight from d_out via gather-GEMM)
    k_gidx_anchor<<<1, 256, 0, stream>>>(a_index, bufGIDX);
    gemm(stream, outH, DIM, P(PI_ATTN + 0), P(PI_ATTN + 1), nullptr, 0,
         bufQ, DIM, bufGIDX, BB * ANUM, DIM, DIM, 0);
    gemm(stream, outH, DIM, P(PI_ATTN + 2), P(PI_ATTN + 3), nullptr, 0,
         bufK, DIM, bufGIDX, BB * ANUM, DIM, DIM, 0);
    gemm(stream, outH, DIM, P(PI_ATTN + 4), P(PI_ATTN + 5), nullptr, 0,
         bufV, DIM, bufGIDX, BB * ANUM, DIM, DIM, 0);
    k_attn<<<BB * 4, ANUM, 0, stream>>>(bufQ, bufK, bufV, bufAO);
    gemm(stream, bufAO, DIM, P(PI_ATTN + 6), P(PI_ATTN + 7), nullptr, 0,
         outA, DIM, nullptr, BB * ANUM, DIM, DIM, 0);
}
